// MoE_36730560315504
// MI455X (gfx1250) — compile-verified, matrix operand order checked by reading, not run on hardware
//
#include <hip/hip_runtime.h>
#include <hip/hip_bf16.h>
#include <math.h>

// ---------------- Problem constants ----------------
#define N_TOK  4096          // 4 * 1024 tokens
#define DMODEL 1024
#define DHEAD  2048
#define NEXP   8
#define SLOTS  (N_TOK * 2)   // top-2 routing -> exactly 2*N slots total

// GEMM tiling: block = 256 threads (8 waves), tile M=128, N=256, Kchunk=32
#define TM 128
#define TN 256
#define NT (TN / 16)         // 16 n-subtiles per wave

// ---------------- WMMA types (CDNA5 / gfx1250, wave32) ----------------
typedef __attribute__((ext_vector_type(16))) __bf16 v16bf;
typedef __attribute__((ext_vector_type(8)))  float  v8f;
typedef __attribute__((ext_vector_type(2)))  __bf16 v2bf;

union Frag16 { v16bf v; uint4 q[2]; };   // 32 bytes = 16 bf16

union BF4 { __bf16 h[4]; uint2 u; };
union BF2 { v2bf v; unsigned int u; };

__device__ __forceinline__ uint2 pack_bf4(float a, float b, float c, float d) {
#if __has_builtin(__builtin_amdgcn_cvt_pk_bf16_f32)
    BF2 lo, hi;
    lo.v = __builtin_amdgcn_cvt_pk_bf16_f32(a, b);
    hi.v = __builtin_amdgcn_cvt_pk_bf16_f32(c, d);
    uint2 r; r.x = lo.u; r.y = hi.u;
    return r;
#else
    BF4 p;
    p.h[0] = (__bf16)a; p.h[1] = (__bf16)b;
    p.h[2] = (__bf16)c; p.h[3] = (__bf16)d;
    return p.u;
#endif
}

__device__ __forceinline__ unsigned short f2bf1(float f) {
    union { __bf16 h; unsigned short s; } c;
    c.h = (__bf16)f;
    return c.s;
}

__device__ __forceinline__ float gelu_exact(float v) {
    return 0.5f * v * (1.0f + erff(v * 0.70710678118654752440f));
}

// ---------------- Kernel 0: zero output + counters ----------------
__global__ __launch_bounds__(256) void k_init(float* __restrict__ y,
                                              int* __restrict__ counts,
                                              int* __restrict__ cursors) {
    int i = blockIdx.x * blockDim.x + threadIdx.x;
    if (blockIdx.x == 0 && threadIdx.x < NEXP) {
        counts[threadIdx.x] = 0;
        cursors[threadIdx.x] = 0;
    }
    const int n4 = (N_TOK * DMODEL) / 4;
    float4 z = make_float4(0.f, 0.f, 0.f, 0.f);
    for (int j = i; j < n4; j += gridDim.x * blockDim.x)
        ((float4*)y)[j] = z;
}

// ---------------- Kernel 1: gating (one wave32 per token) ----------------
__global__ __launch_bounds__(256) void k_gate(const float* __restrict__ x,
                                              const float* __restrict__ gw,
                                              int* __restrict__ counts,
                                              int* __restrict__ t2e,
                                              float* __restrict__ t2g) {
    int wv   = threadIdx.x >> 5;
    int lane = threadIdx.x & 31;
    int t = blockIdx.x * 8 + wv;
    if (t >= N_TOK) return;

    float p[NEXP];
#pragma unroll
    for (int e = 0; e < NEXP; ++e) p[e] = 0.f;

    const float* xr = x + (size_t)t * DMODEL;
    for (int j = lane; j < DMODEL; j += 32) {
        float xv = xr[j];
#pragma unroll
        for (int e = 0; e < NEXP; ++e) p[e] += xv * gw[e * DMODEL + j];
    }
#pragma unroll
    for (int off = 16; off > 0; off >>= 1) {
#pragma unroll
        for (int e = 0; e < NEXP; ++e) p[e] += __shfl_xor(p[e], off, 32);
    }
    if (lane == 0) {
        float m = p[0];
#pragma unroll
        for (int e = 1; e < NEXP; ++e) m = fmaxf(m, p[e]);
        float pr[NEXP], s = 0.f;
#pragma unroll
        for (int e = 0; e < NEXP; ++e) { pr[e] = __expf(p[e] - m); s += pr[e]; }
        float inv = 1.f / s;
#pragma unroll
        for (int e = 0; e < NEXP; ++e) pr[e] *= inv;
        // top-2, stable (lowest index wins ties) like jax.lax.top_k
        int i1 = 0;
#pragma unroll
        for (int e = 1; e < NEXP; ++e) if (pr[e] > pr[i1]) i1 = e;
        int i2 = (i1 == 0) ? 1 : 0;
#pragma unroll
        for (int e = 0; e < NEXP; ++e)
            if (e != i1 && e != i2 && pr[e] > pr[i2]) i2 = e;
        t2e[t * 2 + 0] = i1;  t2g[t * 2 + 0] = pr[i1];
        t2e[t * 2 + 1] = i2;  t2g[t * 2 + 1] = pr[i2];
        atomicAdd(&counts[i1], 1);
        atomicAdd(&counts[i2], 1);
    }
}

// ---------------- Kernel 2: tiny exclusive scan over 8 counts ----------------
__global__ void k_scan(const int* __restrict__ counts, int* __restrict__ offsets) {
    if (threadIdx.x == 0) {
        int a = 0;
        for (int e = 0; e < NEXP; ++e) { offsets[e] = a; a += counts[e]; }
        offsets[NEXP] = a;
    }
}

// ---------------- Kernel 3: scatter tokens into compact expert lists --------
__global__ __launch_bounds__(256) void k_scatter(const int* __restrict__ t2e,
                                                 const float* __restrict__ t2g,
                                                 const int* __restrict__ offsets,
                                                 int* __restrict__ cursors,
                                                 int* __restrict__ idx,
                                                 float* __restrict__ gate) {
    int t = blockIdx.x * blockDim.x + threadIdx.x;
    if (t >= N_TOK) return;
#pragma unroll
    for (int kk = 0; kk < 2; ++kk) {
        int e = t2e[t * 2 + kk];
        int pos = atomicAdd(&cursors[e], 1);
        int s = offsets[e] + pos;
        idx[s]  = t;
        gate[s] = t2g[t * 2 + kk];
    }
}

// ---------------- Kernel 4: h = gelu(x_gathered @ w1[e] + b1[e]) ------------
// grid: (DHEAD/TN = 8, 32 m-tiles, 8 experts), block 256 (8 waves)
// software-pipelined: next K-chunk's global loads issue before current WMMAs
__global__ __launch_bounds__(256) void k_ffn1(const float* __restrict__ x,
                                              const float* __restrict__ w1,
                                              const float* __restrict__ b1,
                                              const int* __restrict__ offsets,
                                              const int* __restrict__ idx,
                                              unsigned short* __restrict__ h) {
    __shared__ __align__(16) unsigned short As[TM][40];  // 80B rows, 16B aligned
    __shared__ __align__(16) unsigned short Bs[TN][40];
    __shared__ int toks[TM];

    const int e   = blockIdx.z;
    const int beg = offsets[e], end = offsets[e + 1];
    const int m0  = beg + blockIdx.y * TM;
    if (m0 >= end) return;
    const int n0  = blockIdx.x * TN;
    const int tid = threadIdx.x;

    for (int r = tid; r < TM; r += 256)
        toks[r] = (m0 + r < end) ? idx[m0 + r] : -1;
    __syncthreads();

    const int wv = tid >> 5, lane = tid & 31, lm = lane & 15;
    const int koff = (lane < 16) ? 0 : 8;            // ISA 16-bit A/B lane layout
    const int ar = wv * 16 + lm;

    // per-thread staging coordinates
    const int rA = tid >> 3, cA = (tid & 7) * 4;     // A: 4 iters, row stride 32
    const int nB = tid & (TN - 1);                   // B: 8 iters
    const int kB = (tid >> 8) * 4;                   // 0 for 256 threads

    v8f acc[NT] = {};

    float4 pA[4];
    float  pB[8][4];

    // prologue: load K-chunk 0
    {
#pragma unroll
        for (int i = 0; i < 4; ++i) {
            int t = toks[rA + 32 * i];
            pA[i] = (t >= 0) ? *(const float4*)&x[(size_t)t * DMODEL + 0 + cA]
                             : make_float4(0.f, 0.f, 0.f, 0.f);
        }
#pragma unroll
        for (int i = 0; i < 8; ++i) {
            int k = kB + 4 * (i & 1) + 8 * (i >> 1) * 0;  // see note below
            (void)k;
        }
        // B: 8 iterations over k = 4*i (n = tid, fixed per thread) -- use lin mapping
#pragma unroll
        for (int i = 0; i < 8; ++i) {
            int lin = tid + 256 * i;
            int n = lin & (TN - 1);
            int k = (lin >> 8) * 4;
            const float* wp = &w1[((size_t)e * DMODEL + 0 + k) * DHEAD + n0 + n];
            pB[i][0] = wp[0 * DHEAD];
            pB[i][1] = wp[1 * DHEAD];
            pB[i][2] = wp[2 * DHEAD];
            pB[i][3] = wp[3 * DHEAD];
        }
    }

    for (int kk = 0; kk < DMODEL; kk += 32) {
        // convert + store staged chunk to LDS
#pragma unroll
        for (int i = 0; i < 4; ++i)
            *(uint2*)&As[rA + 32 * i][cA] = pack_bf4(pA[i].x, pA[i].y, pA[i].z, pA[i].w);
#pragma unroll
        for (int i = 0; i < 8; ++i) {
            int lin = tid + 256 * i;
            int n = lin & (TN - 1);
            int k = (lin >> 8) * 4;
            *(uint2*)&Bs[n][k] = pack_bf4(pB[i][0], pB[i][1], pB[i][2], pB[i][3]);
        }
        __syncthreads();

        // issue next chunk's global loads before the WMMA sequence
        const int kn = kk + 32;
        if (kn < DMODEL) {
#pragma unroll
            for (int i = 0; i < 4; ++i) {
                int t = toks[rA + 32 * i];
                pA[i] = (t >= 0) ? *(const float4*)&x[(size_t)t * DMODEL + kn + cA]
                                 : make_float4(0.f, 0.f, 0.f, 0.f);
            }
#pragma unroll
            for (int i = 0; i < 8; ++i) {
                int lin = tid + 256 * i;
                int n = lin & (TN - 1);
                int k = (lin >> 8) * 4;
                const float* wp = &w1[((size_t)e * DMODEL + kn + k) * DHEAD + n0 + n];
                pB[i][0] = wp[0 * DHEAD];
                pB[i][1] = wp[1 * DHEAD];
                pB[i][2] = wp[2 * DHEAD];
                pB[i][3] = wp[3 * DHEAD];
            }
        }

        Frag16 a;
        a.q[0] = *(const uint4*)&As[ar][koff];       // K = koff .. koff+7
        a.q[1] = *(const uint4*)&As[ar][16 + koff];  // K = 16+koff .. 16+koff+7
#pragma unroll
        for (int nt = 0; nt < NT; ++nt) {
            Frag16 b;
            int br = nt * 16 + lm;
            b.q[0] = *(const uint4*)&Bs[br][koff];
            b.q[1] = *(const uint4*)&Bs[br][16 + koff];
            acc[nt] = __builtin_amdgcn_wmma_f32_16x16x32_bf16(
                false, a.v, false, b.v, (short)0, acc[nt], false, false);
        }
        __syncthreads();
    }

    // epilogue: bias + exact GELU -> bf16 h[slot][col]
    const int mbase = wv * 16 + ((lane < 16) ? 0 : 8);
#pragma unroll
    for (int i = 0; i < 8; ++i) {
        int slot = m0 + mbase + i;
        if (slot < end) {
#pragma unroll
            for (int nt = 0; nt < NT; ++nt) {
                int col = n0 + nt * 16 + lm;
                float v = acc[nt][i] + b1[e * DHEAD + col];
                h[(size_t)slot * DHEAD + col] = f2bf1(gelu_exact(v));
            }
        }
    }
}

// ---------------- Kernel 5: y += gate * (h @ w2[e] + b2[e]) -----------------
// grid: (DMODEL/TN = 4, 32 m-tiles, 8 experts), block 256
__global__ __launch_bounds__(256) void k_ffn2(const unsigned short* __restrict__ h,
                                              const float* __restrict__ w2,
                                              const float* __restrict__ b2,
                                              const int* __restrict__ offsets,
                                              const int* __restrict__ idx,
                                              const float* __restrict__ gate,
                                              float* __restrict__ y) {
    __shared__ __align__(16) unsigned short As[TM][40];
    __shared__ __align__(16) unsigned short Bs[TN][40];
    __shared__ int   toks[TM];
    __shared__ float gts[TM];

    const int e   = blockIdx.z;
    const int beg = offsets[e], end = offsets[e + 1];
    const int m0  = beg + blockIdx.y * TM;
    if (m0 >= end) return;
    const int n0  = blockIdx.x * TN;
    const int tid = threadIdx.x;

    for (int r = tid; r < TM; r += 256) {
        bool ok = (m0 + r < end);
        toks[r] = ok ? idx[m0 + r] : -1;
        gts[r]  = ok ? gate[m0 + r] : 0.f;
    }
    __syncthreads();

    const int wv = tid >> 5, lane = tid & 31, lm = lane & 15;
    const int koff = (lane < 16) ? 0 : 8;
    const int ar = wv * 16 + lm;

    // staging coordinates: A = bf16 copy (2 x uint4), B = transpose-convert
    const int rA = tid >> 2, qA = (tid & 3) * 8;

    v8f acc[NT] = {};

    uint4 pA[2];
    float pB[8][4];

    // prologue: load K-chunk 0
    {
#pragma unroll
        for (int i = 0; i < 2; ++i) {
            int slot = m0 + rA + 64 * i;
            pA[i] = (slot < end) ? *(const uint4*)&h[(size_t)slot * DHEAD + 0 + qA]
                                 : make_uint4(0u, 0u, 0u, 0u);
        }
#pragma unroll
        for (int i = 0; i < 8; ++i) {
            int lin = tid + 256 * i;
            int n = lin & (TN - 1);
            int k = (lin >> 8) * 4;
            const float* wp = &w2[((size_t)e * DHEAD + 0 + k) * DMODEL + n0 + n];
            pB[i][0] = wp[0 * DMODEL];
            pB[i][1] = wp[1 * DMODEL];
            pB[i][2] = wp[2 * DMODEL];
            pB[i][3] = wp[3 * DMODEL];
        }
    }

    for (int kk = 0; kk < DHEAD; kk += 32) {
        // store staged chunk to LDS
#pragma unroll
        for (int i = 0; i < 2; ++i)
            *(uint4*)&As[rA + 64 * i][qA] = pA[i];
#pragma unroll
        for (int i = 0; i < 8; ++i) {
            int lin = tid + 256 * i;
            int n = lin & (TN - 1);
            int k = (lin >> 8) * 4;
            *(uint2*)&Bs[n][k] = pack_bf4(pB[i][0], pB[i][1], pB[i][2], pB[i][3]);
        }
        __syncthreads();

        // issue next chunk's global loads before the WMMA sequence
        const int kn = kk + 32;
        if (kn < DHEAD) {
#pragma unroll
            for (int i = 0; i < 2; ++i) {
                int slot = m0 + rA + 64 * i;
                pA[i] = (slot < end) ? *(const uint4*)&h[(size_t)slot * DHEAD + kn + qA]
                                     : make_uint4(0u, 0u, 0u, 0u);
            }
#pragma unroll
            for (int i = 0; i < 8; ++i) {
                int lin = tid + 256 * i;
                int n = lin & (TN - 1);
                int k = (lin >> 8) * 4;
                const float* wp = &w2[((size_t)e * DHEAD + kn + k) * DMODEL + n0 + n];
                pB[i][0] = wp[0 * DMODEL];
                pB[i][1] = wp[1 * DMODEL];
                pB[i][2] = wp[2 * DMODEL];
                pB[i][3] = wp[3 * DMODEL];
            }
        }

        Frag16 a;
        a.q[0] = *(const uint4*)&As[ar][koff];
        a.q[1] = *(const uint4*)&As[ar][16 + koff];
#pragma unroll
        for (int nt = 0; nt < NT; ++nt) {
            Frag16 b;
            int br = nt * 16 + lm;
            b.q[0] = *(const uint4*)&Bs[br][koff];
            b.q[1] = *(const uint4*)&Bs[br][16 + koff];
            acc[nt] = __builtin_amdgcn_wmma_f32_16x16x32_bf16(
                false, a.v, false, b.v, (short)0, acc[nt], false, false);
        }
        __syncthreads();
    }

    // epilogue: scale by gate, add bias, atomic accumulate into y[token]
    // each y element receives exactly 2 atomic adds (commutative -> deterministic)
    const int mbase = wv * 16 + ((lane < 16) ? 0 : 8);
#pragma unroll
    for (int i = 0; i < 8; ++i) {
        int slot = m0 + mbase + i;
        if (slot < end) {
            int tok = toks[mbase + i];
            float g = gts[mbase + i];
#pragma unroll
            for (int nt = 0; nt < NT; ++nt) {
                int col = n0 + nt * 16 + lm;
                float v = g * (acc[nt][i] + b2[e * DMODEL + col]);
                atomicAdd(&y[(size_t)tok * DMODEL + col], v);
            }
        }
    }
}

// ---------------- Host launcher ----------------
extern "C" void kernel_launch(void* const* d_in, const int* in_sizes, int n_in,
                              void* d_out, int out_size, void* d_ws, size_t ws_size,
                              hipStream_t stream) {
    const float* x  = (const float*)d_in[0];
    const float* gw = (const float*)d_in[1];
    const float* w1 = (const float*)d_in[2];
    const float* b1 = (const float*)d_in[3];
    const float* w2 = (const float*)d_in[4];
    const float* b2 = (const float*)d_in[5];
    float* y = (float*)d_out;

    // workspace layout (bytes):
    //   [0,64) counts[8] | [64,128) cursors[8] | [128,256) offsets[9]
    //   [256,+32KB) t2e | +32KB t2g | +32KB idx | +32KB gate
    //   then h [SLOTS * DHEAD] bf16 (33.5 MB)
    char* ws = (char*)d_ws;
    int*   counts  = (int*)  (ws + 0);
    int*   cursors = (int*)  (ws + 64);
    int*   offsets = (int*)  (ws + 128);
    int*   t2e     = (int*)  (ws + 256);
    float* t2g     = (float*)(ws + 256 + (size_t)SLOTS * 4);
    int*   idx     = (int*)  (ws + 256 + (size_t)SLOTS * 8);
    float* gate    = (float*)(ws + 256 + (size_t)SLOTS * 12);
    unsigned short* h = (unsigned short*)(ws + 256 + (size_t)SLOTS * 16);

    k_init   <<<256,            256, 0, stream>>>(y, counts, cursors);
    k_gate   <<<N_TOK / 8,      256, 0, stream>>>(x, gw, counts, t2e, t2g);
    k_scan   <<<1,               32, 0, stream>>>(counts, offsets);
    k_scatter<<<N_TOK / 256,    256, 0, stream>>>(t2e, t2g, offsets, cursors, idx, gate);
    k_ffn1   <<<dim3(DHEAD  / TN, 32, NEXP), 256, 0, stream>>>(x, w1, b1, offsets, idx, h);
    k_ffn2   <<<dim3(DMODEL / TN, 32, NEXP), 256, 0, stream>>>(h, w2, b2, offsets, idx, gate, y);
}
